// SUPRAParallelFixedGamma_10024453668953
// MI455X (gfx1250) — matrix operand order repaired and checked by reading, not compiled, not run on hardware
//
#include <hip/hip_runtime.h>
#include <hip/hip_bf16.h>

// ---------------------------------------------------------------------------
// SUPRAParallelFixedGamma on MI455X (gfx1250), algebraically reduced:
//   QK[b,t,j]  = sum_d q[b,t,d] k[b,j,d]
//   out[b,t,j] = C[t,j]*QK[b,t,j]*vd[b,j] / (sum_j' C[t,j']*QK[b,t,j'] + eps)
// C[t,j] in closed form (no cumsum, no Z GEMM).
//
// All GEMMs run on v_wmma_f32_16x16x32_bf16 with a bf16 hi/lo (bf16x3) split.
// Operands are pre-split into bf16 planes ONCE, so the GEMM hot loop moves
// pure bf16 bytes with GLOBAL_LOAD_ASYNC_TO_LDS_B128 (ASYNCcnt-tracked,
// double-buffered) -- no VGPR staging, no per-tile conversion VALU.
// ---------------------------------------------------------------------------

typedef __attribute__((ext_vector_type(16))) __bf16 v16bf;
typedef __attribute__((ext_vector_type(8)))  float  v8f;

struct alignas(16) U4 { unsigned x, y, z, w; };
struct alignas(8)  U2 { unsigned x, y; };
struct alignas(16) F4 { float x, y, z, w; };

union FragU { v16bf v; U4 q[2]; };

__device__ __forceinline__ v8f wmma_bf16(v16bf a, v16bf b, v8f c) {
  return __builtin_amdgcn_wmma_f32_16x16x32_bf16(
      false, a, false, b, (short)0, c, false, false);
}

// Split two fp32 into packed bf16 hi (truncated top-16) and bf16 lo (residual).
__device__ __forceinline__ void split2(float a, float b, unsigned& hi, unsigned& lo) {
  unsigned ua = __float_as_uint(a), ub = __float_as_uint(b);
  hi = (ua >> 16) | (ub & 0xFFFF0000u);
  float ra = a - __uint_as_float(ua & 0xFFFF0000u);
  float rb = b - __uint_as_float(ub & 0xFFFF0000u);
  lo = (__float_as_uint(ra) >> 16) | (__float_as_uint(rb) & 0xFFFF0000u);
}

// Async 16-byte copy global -> LDS (GV mode: vdst = LDS byte offset,
// vaddr = 64-bit global address, saddr = off). Tracked with ASYNCcnt.
__device__ __forceinline__ void async_copy_b128(const unsigned short* g, unsigned ldsOff) {
  asm volatile("global_load_async_to_lds_b128 %0, %1, off"
               :: "v"(ldsOff), "v"(g) : "memory");
}
__device__ __forceinline__ unsigned lds_off(const void* p) {
  return (unsigned)(size_t)p;   // flat LDS addr low 32 bits = wave LDS offset
}

#define TBK   32
#define STR   40                 // padded ushort stride per LDS tile row
#define PLANE (128 * STR)        // one 128x32 bf16 tile plane (in ushorts)

// ---------------------------------------------------------------------------
// D(z) = A(z) [MxK] * B(z)^T [NxK], A/B given as bf16 hi/lo planes, K-major.
// Output: either f32 (OutF) or split bf16 hi/lo planes (OutH/OutL), +bias/relu.
// 128x128 tile per 256-thread WG (8 waves: 4 along M x 2 along N),
// double-buffered async LDS fills.
// ---------------------------------------------------------------------------
__global__ __launch_bounds__(256)
void gemm_nt_bf16x3_async(const unsigned short* __restrict__ AH,
                          const unsigned short* __restrict__ AL, int lda, long long sAz,
                          const unsigned short* __restrict__ BH,
                          const unsigned short* __restrict__ BL, int ldb, long long sBz,
                          float* __restrict__ OutF,
                          unsigned short* __restrict__ OutH,
                          unsigned short* __restrict__ OutL, int ldo, long long sOz,
                          int K, const float* __restrict__ bias, int relu)
{
  __shared__ unsigned short LDS[2 * 4 * PLANE];   // 2 buffers x {AH,AL,BH,BL}

  AH += (long long)blockIdx.z * sAz;  AL += (long long)blockIdx.z * sAz;
  BH += (long long)blockIdx.z * sBz;  BL += (long long)blockIdx.z * sBz;

  const int mBase = blockIdx.y * 128;
  const int nBase = blockIdx.x * 128;

  const int tid  = threadIdx.x;
  const int lane = tid & 31;
  const int wid  = tid >> 5;
  const int wm   = wid & 3;   // 32 rows per wave
  const int wn   = wid >> 2;  // 64 cols per wave
  const int sub  = lane >> 4;
  const int r    = lane & 15;

  const v8f vzero = {0.f, 0.f, 0.f, 0.f, 0.f, 0.f, 0.f, 0.f};
  v8f acc[2][4];
#pragma unroll
  for (int mi = 0; mi < 2; ++mi)
#pragma unroll
    for (int ni = 0; ni < 4; ++ni) acc[mi][ni] = vzero;

  // Per-thread fill of one K-step buffer: 2 chunks x 4 planes, 16B each.
  auto fill = [&](int buf, int k0) {
#pragma unroll
    for (int it = 0; it < 2; ++it) {
      int c    = it * 256 + tid;
      int row  = c >> 2;            // 4 x 8-half chunks per 32-wide row
      int col8 = (c & 3) << 3;
      unsigned lo0 = lds_off(&LDS[buf * 4 * PLANE + row * STR + col8]);
      size_t ao = (size_t)(mBase + row) * lda + k0 + col8;
      size_t bo = (size_t)(nBase + row) * ldb + k0 + col8;
      async_copy_b128(AH + ao, lo0);
      async_copy_b128(AL + ao, lo0 + 2 * PLANE);
      async_copy_b128(BH + bo, lo0 + 4 * PLANE);
      async_copy_b128(BL + bo, lo0 + 6 * PLANE);
    }
  };

  const int KT = K / TBK;
  fill(0, 0);
  for (int kt = 0; kt < KT; ++kt) {
    const int cur = kt & 1;
    if (kt + 1 < KT) {
      fill(cur ^ 1, (kt + 1) * TBK);                 // prefetch next tile
      asm volatile("s_wait_asynccnt 0x8" ::: "memory");  // cur's 8 done (in-order)
    } else {
      asm volatile("s_wait_asynccnt 0x0" ::: "memory");
    }
    __syncthreads();                                  // all waves' fills visible

    const unsigned short* sAH = &LDS[cur * 4 * PLANE + 0 * PLANE];
    const unsigned short* sAL = &LDS[cur * 4 * PLANE + 1 * PLANE];
    const unsigned short* sBH = &LDS[cur * 4 * PLANE + 2 * PLANE];
    const unsigned short* sBL = &LDS[cur * 4 * PLANE + 3 * PLANE];

    FragU aH[2], aL[2];
#pragma unroll
    for (int mi = 0; mi < 2; ++mi) {
      int base = (wm * 32 + mi * 16 + r) * STR;
      aH[mi].q[0] = *(const U4*)&sAH[base + sub * 8];
      aH[mi].q[1] = *(const U4*)&sAH[base + 16 + sub * 8];
      aL[mi].q[0] = *(const U4*)&sAL[base + sub * 8];
      aL[mi].q[1] = *(const U4*)&sAL[base + 16 + sub * 8];
    }
#pragma unroll
    for (int ni = 0; ni < 4; ++ni) {
      int bbase = (wn * 64 + ni * 16 + r) * STR + sub * 8;
      FragU bH, bL;
      bH.q[0] = *(const U4*)&sBH[bbase];
      bH.q[1] = *(const U4*)&sBH[bbase + 16];
      bL.q[0] = *(const U4*)&sBL[bbase];
      bL.q[1] = *(const U4*)&sBL[bbase + 16];
#pragma unroll
      for (int mi = 0; mi < 2; ++mi) {
        v8f c = acc[mi][ni];
        c = wmma_bf16(aL[mi].v, bH.v, c);   // lo*hi
        c = wmma_bf16(aH[mi].v, bL.v, c);   // hi*lo
        c = wmma_bf16(aH[mi].v, bH.v, c);   // hi*hi
        acc[mi][ni] = c;
      }
    }
    __syncthreads();   // all waves done reading `cur` before it is refilled
  }

  // C/D layout: VGPR g -> M = sub*8 + g, N = r within each 16x16 tile.
#pragma unroll
  for (int ni = 0; ni < 4; ++ni) {
    int gn = nBase + wn * 64 + ni * 16 + r;
    float bb = bias ? bias[gn] : 0.0f;
#pragma unroll
    for (int mi = 0; mi < 2; ++mi) {
      int gm0 = mBase + wm * 32 + mi * 16 + sub * 8;
#pragma unroll
      for (int g = 0; g < 8; ++g) {
        float val = acc[mi][ni][g] + bb;
        if (relu) val = fmaxf(val, 0.0f);
        size_t o = (size_t)(gm0 + g) * ldo + gn;
        if (OutF) {
          OutF[o] = val;
        } else {
          unsigned u = __float_as_uint(val);
          OutH[o] = (unsigned short)(u >> 16);
          float rem = val - __uint_as_float(u & 0xFFFF0000u);
          OutL[o] = (unsigned short)(__float_as_uint(rem) >> 16);
        }
      }
    }
  }
}

// Elementwise fp32 -> bf16 hi/lo planes (one-time pre-split, float4 per thread).
__global__ __launch_bounds__(256)
void split_f32_kernel(const float* __restrict__ in,
                      unsigned short* __restrict__ h,
                      unsigned short* __restrict__ l)
{
  size_t i = ((size_t)blockIdx.x * 256 + threadIdx.x) * 4;
  F4 v = *(const F4*)(in + i);
  unsigned h01, l01, h23, l23;
  split2(v.x, v.y, h01, l01);
  split2(v.z, v.w, h23, l23);
  *(U2*)(h + i) = U2{h01, h23};
  *(U2*)(l + i) = U2{l01, l23};
}

// vd[b*T + j] = dot(x[b,j,:], Wv[j,:]) + bv[j]  -- one wave per row, fp32 exact.
__global__ __launch_bounds__(256)
void vdiag_kernel(const float* __restrict__ x, const float* __restrict__ Wv,
                  const float* __restrict__ bv, float* __restrict__ vd,
                  int T, int D)
{
  int row  = blockIdx.x * 8 + (threadIdx.x >> 5);
  int lane = threadIdx.x & 31;
  int j    = row % T;
  const float* xr = x  + (size_t)row * D;
  const float* wr = Wv + (size_t)j * D;
  float s = 0.0f;
  for (int i = lane; i < D; i += 32) s = fmaf(xr[i], wr[i], s);
#pragma unroll
  for (int off = 16; off > 0; off >>= 1) s += __shfl_xor(s, off, 32);
  if (lane == 0) vd[row] = s + bv[j];
}

// One block per (b,t) row of QK (in d_out): analytic C[t,j], denom reduce,
// in-place rescale.
__global__ __launch_bounds__(256)
void epilogue_kernel(float* __restrict__ out, const float* __restrict__ vd,
                     const float* __restrict__ gamma_p, int T)
{
  __shared__ float red[256];
  const int row = blockIdx.x;           // b*T + t
  const int t   = row % T;
  const int b   = row / T;
  const float gamma = *gamma_p;
  const float log2g = log2f(gamma);
  const float rinv  = 1.0f / (1.0f - gamma);
  float* orow = out + (size_t)row * T;
  const float* vdb = vd + (size_t)b * T;

  float w[4], vv[4], part = 0.0f;
#pragma unroll
  for (int i = 0; i < 4; ++i) {
    int j = threadIdx.x + i * 256;
    float c;
    if (j >= t)
      c = exp2f((float)(j - t) * log2g) *
          (1.0f - exp2f((float)(t + 1) * log2g)) * rinv;
    else
      c = (1.0f - exp2f((float)(j + 1) * log2g)) * rinv + (float)(t - j);
    w[i]  = c * orow[j];
    vv[i] = vdb[j];
    part += w[i];
  }
  red[threadIdx.x] = part;
  __syncthreads();
  for (int s = 128; s > 0; s >>= 1) {
    if (threadIdx.x < (unsigned)s) red[threadIdx.x] += red[threadIdx.x + s];
    __syncthreads();
  }
  const float inv = 1.0f / (red[0] + 1e-6f);
#pragma unroll
  for (int i = 0; i < 4; ++i) {
    int j = threadIdx.x + i * 256;
    orow[j] = w[i] * vv[i] * inv;
  }
}

extern "C" void kernel_launch(void* const* d_in, const int* in_sizes, int n_in,
                              void* d_out, int out_size, void* d_ws, size_t ws_size,
                              hipStream_t stream) {
  const float* x     = (const float*)d_in[0];
  const float* gamma = (const float*)d_in[1];
  const float* Wq    = (const float*)d_in[2];
  const float* bq    = (const float*)d_in[3];
  const float* Wk    = (const float*)d_in[4];
  const float* bk    = (const float*)d_in[5];
  const float* Wv    = (const float*)d_in[6];
  const float* bv    = (const float*)d_in[7];
  float* out = (float*)d_out;

  const int T = 1024, D = 1024;
  const int Bn = in_sizes[0] / (T * D);   // 4
  const int M  = Bn * T;                  // 4096
  const size_t MD = (size_t)M * D, DD = (size_t)D * D;

  unsigned short* p = (unsigned short*)d_ws;
  unsigned short *xH = p, *xL = p + MD;            p += 2 * MD;
  unsigned short *WqH = p, *WqL = p + DD;          p += 2 * DD;
  unsigned short *WkH = p, *WkL = p + DD;          p += 2 * DD;
  unsigned short *qH = p, *qL = p + MD;            p += 2 * MD;
  unsigned short *kH = p, *kL = p + MD;            p += 2 * MD;
  float* vd = (float*)p;

  dim3 blk(256);

  // One-time fp32 -> bf16 hi/lo splits.
  split_f32_kernel<<<dim3((unsigned)(MD / 1024)), blk, 0, stream>>>(x, xH, xL);
  split_f32_kernel<<<dim3((unsigned)(DD / 1024)), blk, 0, stream>>>(Wq, WqH, WqL);
  split_f32_kernel<<<dim3((unsigned)(DD / 1024)), blk, 0, stream>>>(Wk, WkH, WkL);

  // q = relu(x Wq^T + bq), k = relu(x Wk^T + bk): emitted directly as hi/lo.
  gemm_nt_bf16x3_async<<<dim3(D / 128, M / 128, 1), blk, 0, stream>>>(
      xH, xL, D, 0LL, WqH, WqL, D, 0LL, nullptr, qH, qL, D, 0LL, D, bq, 1);
  gemm_nt_bf16x3_async<<<dim3(D / 128, M / 128, 1), blk, 0, stream>>>(
      xH, xL, D, 0LL, WkH, WkL, D, 0LL, nullptr, kH, kL, D, 0LL, D, bk, 1);

  // vd[b,j] = (x[b,j,:] . Wv[j,:]) + bv[j]  (fp32 exact, tiny)
  vdiag_kernel<<<dim3(M / 8), blk, 0, stream>>>(x, Wv, bv, vd, T, D);

  // QK[b] = q[b] k[b]^T, straight into d_out (f32).
  gemm_nt_bf16x3_async<<<dim3(T / 128, T / 128, Bn), blk, 0, stream>>>(
      qH, qL, D, (long long)T * D, kH, kL, D, (long long)T * D,
      out, nullptr, nullptr, T, (long long)T * T, D, nullptr, 0);

  // out = C*QK*vd / (sum_j C*QK + eps), in place.
  epilogue_kernel<<<dim3(M), blk, 0, stream>>>(out, vd, gamma, T);
}
